// AttentionModel_64630667870790
// MI455X (gfx1250) — matrix operand order repaired and checked by reading, not compiled, not run on hardware
//
#include <hip/hip_runtime.h>
#include <math.h>

// ---------------------------------------------------------------------------
// Types for gfx1250 WMMA (bf16 inputs, f32 accumulate)
// ---------------------------------------------------------------------------
typedef __attribute__((ext_vector_type(16))) __bf16 v16bf;
typedef __attribute__((ext_vector_type(8)))  float  v8f;

#define DMODEL 512
#define NHEAD  16
#define NMEM   251
#define TSA    256
#define BATCH  512

// ---------------------------------------------------------------------------
// GEMM: C[B x 512] = A[B x 512] @ W[512 x 512] + bias (+ resid) (optional ReLU)
// One wave (32 lanes) computes one 16x16 output tile via v_wmma_f32_16x16x32_bf16.
// A packing used here (contiguous k, matched with B): lane half h, element j -> K = 16*h + j
// B packing used here (contiguous k, matched with A): lane half h, element j -> K = 16*h + j
// C/D layout: VGPR r, lane l -> M = r + 8*(l>>4), N = (l&15)
// Compile-time RESID/RELU variants keep the epilogue branch-free.
// ---------------------------------------------------------------------------
template <bool HAS_RESID, bool HAS_RELU>
__global__ __launch_bounds__(32)
void gemm512_wmma(const float* __restrict__ A, const float* __restrict__ W,
                  const float* __restrict__ bias, const float* __restrict__ resid,
                  float* __restrict__ C)
{
    const int tn   = blockIdx.x;          // column tile
    const int tm   = blockIdx.y;          // row tile
    const int lane = threadIdx.x;         // 0..31
    const int half = lane >> 4;
    const int mrow = (tm << 4) + (lane & 15);
    const int ncol = (tn << 4) + (lane & 15);

    v8f acc = {};
#pragma unroll 4
    for (int kb = 0; kb < DMODEL; kb += 32) {
        v16bf a, b;
#pragma unroll
        for (int j = 0; j < 16; ++j) {
            const int ka = kb + (half << 4) + j;
            const int kw = kb + (half << 4) + j;
            a[j] = (__bf16)A[(size_t)mrow * DMODEL + ka];
            b[j] = (__bf16)W[(size_t)kw   * DMODEL + ncol];
        }
        acc = __builtin_amdgcn_wmma_f32_16x16x32_bf16(
                  false, a, false, b, (short)0, acc, false, false);
    }

    const float bn = bias[ncol];
#pragma unroll
    for (int r = 0; r < 8; ++r) {
        const int m = (tm << 4) + r + (half << 3);
        float v = acc[r] + bn;
        if (HAS_RESID) v += resid[(size_t)m * DMODEL + ncol];
        if (HAS_RELU)  v = v > 0.f ? v : 0.f;
        C[(size_t)m * DMODEL + ncol] = v;
    }
}

// ---------------------------------------------------------------------------
// LayerNorm over last dim (512). One block per row, 256 threads.
// ---------------------------------------------------------------------------
__global__ __launch_bounds__(256)
void layernorm512(const float* __restrict__ x, const float* __restrict__ g,
                  const float* __restrict__ beta, float* __restrict__ out)
{
    const int b = blockIdx.x, tid = threadIdx.x;
    __shared__ float red[256];
    const float v0 = x[(size_t)b * DMODEL + tid];
    const float v1 = x[(size_t)b * DMODEL + tid + 256];
    red[tid] = v0 + v1;
    __syncthreads();
    for (int s = 128; s > 0; s >>= 1) { if (tid < s) red[tid] += red[tid + s]; __syncthreads(); }
    const float mean = red[0] * (1.f / DMODEL);
    __syncthreads();
    const float d0 = v0 - mean, d1 = v1 - mean;
    red[tid] = d0 * d0 + d1 * d1;
    __syncthreads();
    for (int s = 128; s > 0; s >>= 1) { if (tid < s) red[tid] += red[tid + s]; __syncthreads(); }
    const float rstd = rsqrtf(red[0] * (1.f / DMODEL) + 1e-5f);
    out[(size_t)b * DMODEL + tid]       = g[tid]       * d0 * rstd + beta[tid];
    out[(size_t)b * DMODEL + tid + 256] = g[tid + 256] * d1 * rstd + beta[tid + 256];
}

// ---------------------------------------------------------------------------
// Multi-head attention (16 heads, dh=32) for a single query per batch row.
// One block per batch row, 256 threads (thread t owns d = 2t, 2t+1; head = t>>4).
// Streams each K/V row from HBM exactly once, fully coalesced.
// kx/vx: optional extra (appended) key/value row of length 512 (self-attn).
// ---------------------------------------------------------------------------
#define MAXK 257
__global__ __launch_bounds__(256)
void mha_qkv(const float* __restrict__ q, const float* __restrict__ Kb,
             const float* __restrict__ Vb, const float* __restrict__ kx,
             const float* __restrict__ vx, const int* __restrict__ mask,
             int nbase, int nkeys, int rowstride, float scale,
             float* __restrict__ out)
{
    const int b = blockIdx.x, tid = threadIdx.x;
    const int h = tid >> 4;
    __shared__ float sc[NHEAD][MAXK + 3];
    __shared__ float qsh[DMODEL];

    qsh[tid]       = q[(size_t)b * DMODEL + tid];
    qsh[tid + 256] = q[(size_t)b * DMODEL + tid + 256];
    __syncthreads();
    const float2 qv = ((const float2*)qsh)[tid];

    // Phase 1: scores[h][k], coalesced row streaming + 16-lane shuffle reduce
    for (int k = 0; k < nkeys; ++k) {
        const float* row = (k < nbase) ? (Kb + ((size_t)b * nbase + k) * rowstride)
                                       : (kx + (size_t)b * DMODEL);
        if (k + 1 < nbase)
            __builtin_prefetch(Kb + ((size_t)b * nbase + k + 1) * rowstride + 2 * tid, 0, 0);
        const float2 rv = ((const float2*)row)[tid];
        float p = rv.x * qv.x + rv.y * qv.y;
        p += __shfl_xor(p, 8);
        p += __shfl_xor(p, 4);
        p += __shfl_xor(p, 2);
        p += __shfl_xor(p, 1);
        if ((tid & 15) == 0) {
            float s = p * scale;
            if (mask && mask[(size_t)b * nbase + k] != 0) s = -1.0e9f;
            sc[h][k] = s;
        }
    }
    __syncthreads();

    // Phase 2: softmax per head (cheap: LDS-resident)
    if (tid < NHEAD) {
        float m = -INFINITY;
        for (int k = 0; k < nkeys; ++k) m = fmaxf(m, sc[tid][k]);
        float sum = 0.f;
        for (int k = 0; k < nkeys; ++k) { float e = __expf(sc[tid][k] - m); sc[tid][k] = e; sum += e; }
        const float inv = 1.f / sum;
        for (int k = 0; k < nkeys; ++k) sc[tid][k] *= inv;
    }
    __syncthreads();

    // Phase 3: out = sum_k w[h][k] * V[k][:], coalesced row streaming
    float a0 = 0.f, a1 = 0.f;
    for (int k = 0; k < nkeys; ++k) {
        const float* row = (k < nbase) ? (Vb + ((size_t)b * nbase + k) * rowstride)
                                       : (vx + (size_t)b * DMODEL);
        if (k + 1 < nbase)
            __builtin_prefetch(Vb + ((size_t)b * nbase + k + 1) * rowstride + 2 * tid, 0, 0);
        const float2 rv = ((const float2*)row)[tid];
        const float w = sc[h][k];
        a0 += w * rv.x;
        a1 += w * rv.y;
    }
    out[(size_t)b * DMODEL + 2 * tid]     = a0;
    out[(size_t)b * DMODEL + 2 * tid + 1] = a1;
}

// ---------------------------------------------------------------------------
// Final single-head attention: returns softmax weights only (B x 251).
// scores = 10*tanh(dot(qf, K_att[:, :, 512:]) / sqrt(512)), masked.
// One block per batch row; thread t owns key t (contiguous 2KB row read).
// ---------------------------------------------------------------------------
__global__ __launch_bounds__(256)
void final_attn(const float* __restrict__ qf, const float* __restrict__ Katt,
                const int* __restrict__ mask, float* __restrict__ out)
{
    const int b = blockIdx.x, tid = threadIdx.x;
    __shared__ float qsh[DMODEL];
    __shared__ float sc[256];
    __shared__ float red[256];

    qsh[tid]       = qf[(size_t)b * DMODEL + tid];
    qsh[tid + 256] = qf[(size_t)b * DMODEL + tid + 256];
    __syncthreads();

    float s = -INFINITY;
    if (tid < NMEM) {
        const float4* rp = (const float4*)(Katt + ((size_t)b * NMEM + tid) * 1024 + 512);
        const float4* qp = (const float4*)qsh;
        float dot = 0.f;
#pragma unroll 4
        for (int i = 0; i < 128; ++i) {
            const float4 kv = rp[i], qv = qp[i];
            dot += kv.x * qv.x + kv.y * qv.y + kv.z * qv.z + kv.w * qv.w;
        }
        s = 10.0f * tanhf(dot * 0.044194173824159216f); // 1/sqrt(512)
        if (mask[(size_t)b * NMEM + tid] != 0) s = -1.0e9f;
    }
    sc[tid] = s;
    red[tid] = s;
    __syncthreads();
    for (int st = 128; st > 0; st >>= 1) { if (tid < st) red[tid] = fmaxf(red[tid], red[tid + st]); __syncthreads(); }
    const float m = red[0];
    __syncthreads();
    const float e = (tid < NMEM) ? __expf(sc[tid] - m) : 0.f;
    red[tid] = e;
    __syncthreads();
    for (int st = 128; st > 0; st >>= 1) { if (tid < st) red[tid] += red[tid + st]; __syncthreads(); }
    const float inv = 1.f / red[0];
    if (tid < NMEM) out[(size_t)b * NMEM + tid] = e * inv;
}

// ---------------------------------------------------------------------------
// Host-side orchestration (all launches on `stream`, graph-capture safe)
// ---------------------------------------------------------------------------
extern "C" void kernel_launch(void* const* d_in, const int* in_sizes, int n_in,
                              void* d_out, int out_size, void* d_ws, size_t ws_size,
                              hipStream_t stream)
{
    (void)in_sizes; (void)n_in; (void)out_size; (void)ws_size;
    const float* h_t   = (const float*)d_in[0];
    const float* K_att = (const float*)d_in[1];
    const float* V_att = (const float*)d_in[2];
    const float* K_sa  = (const float*)d_in[3];
    const float* V_sa  = (const float*)d_in[4];
    const float* Wq_sa = (const float*)d_in[5];
    const float* Wk_sa = (const float*)d_in[6];
    const float* Wv_sa = (const float*)d_in[7];
    const float* W0_sa = (const float*)d_in[8];
    const float* Wq_a  = (const float*)d_in[9];
    const float* W0_a  = (const float*)d_in[10];
    const float* W1    = (const float*)d_in[11];
    const float* W2    = (const float*)d_in[12];
    const float* Wqf   = (const float*)d_in[13];
    const float* bq_sa = (const float*)d_in[14];
    const float* bk_sa = (const float*)d_in[15];
    const float* bv_sa = (const float*)d_in[16];
    const float* b0_sa = (const float*)d_in[17];
    const float* bq_a  = (const float*)d_in[18];
    const float* b0_a  = (const float*)d_in[19];
    const float* b1    = (const float*)d_in[20];
    const float* b2    = (const float*)d_in[21];
    const float* bqf   = (const float*)d_in[22];
    const float* ln1_g = (const float*)d_in[23];
    const float* ln2_g = (const float*)d_in[24];
    const float* ln3_g = (const float*)d_in[25];
    const float* ln1_b = (const float*)d_in[26];
    const float* ln2_b = (const float*)d_in[27];
    const float* ln3_b = (const float*)d_in[28];
    const int*   mask  = (const int*)d_in[29];
    float* out = (float*)d_out;

    const size_t S = (size_t)BATCH * DMODEL;     // 512*512 floats = 1 MB
    float* ws = (float*)d_ws;
    float* buf0 = ws + 0 * S;   // q_sa -> q_a -> x3
    float* buf1 = ws + 1 * S;   // k_sa -> x2
    float* buf2 = ws + 2 * S;   // v_sa -> hid
    float* buf3 = ws + 3 * S;   // sa_out -> a_out -> qf
    float* buf4 = ws + 4 * S;   // t1 -> t2 -> t3
    float* buf5 = ws + 5 * S;   // x1

    const dim3 gg(DMODEL / 16, BATCH / 16);      // 32 x 32 tiles, 1 wave each
    const float sc_h  = 0.17677669529663689f;    // 1/sqrt(32)

    // q/k/v projections
    gemm512_wmma<false, false><<<gg, 32, 0, stream>>>(h_t, Wq_sa, bq_sa, nullptr, buf0);
    gemm512_wmma<false, false><<<gg, 32, 0, stream>>>(h_t, Wk_sa, bk_sa, nullptr, buf1);
    gemm512_wmma<false, false><<<gg, 32, 0, stream>>>(h_t, Wv_sa, bv_sa, nullptr, buf2);
    // self-attention over [K_sa ; k_sa] (257 keys)
    mha_qkv<<<BATCH, 256, 0, stream>>>(buf0, K_sa, V_sa, buf1, buf2, nullptr,
                                       TSA, TSA + 1, DMODEL, sc_h, buf3);
    // x1 = LN(h_t + sa_out @ W0_sa + b0_sa)
    gemm512_wmma<true, false><<<gg, 32, 0, stream>>>(buf3, W0_sa, b0_sa, h_t, buf4);
    layernorm512<<<BATCH, 256, 0, stream>>>(buf4, ln1_g, ln1_b, buf5);
    // cross-attention over K_att[:, :, :512] with mask
    gemm512_wmma<false, false><<<gg, 32, 0, stream>>>(buf5, Wq_a, bq_a, nullptr, buf0);
    mha_qkv<<<BATCH, 256, 0, stream>>>(buf0, K_att, V_att, nullptr, nullptr, mask,
                                       NMEM, NMEM, 2 * DMODEL, sc_h, buf3);
    // x2 = LN(x1 + a_out @ W0_a + b0_a)
    gemm512_wmma<true, false><<<gg, 32, 0, stream>>>(buf3, W0_a, b0_a, buf5, buf4);
    layernorm512<<<BATCH, 256, 0, stream>>>(buf4, ln2_g, ln2_b, buf1);
    // FFN: x3 = LN(x2 + relu(x2@W1+b1)@W2 + b2)
    gemm512_wmma<false, true ><<<gg, 32, 0, stream>>>(buf1, W1, b1, nullptr, buf2);
    gemm512_wmma<true, false><<<gg, 32, 0, stream>>>(buf2, W2, b2, buf1, buf4);
    layernorm512<<<BATCH, 256, 0, stream>>>(buf4, ln3_g, ln3_b, buf0);
    // qf and final pointer weights over K_att[:, :, 512:]
    gemm512_wmma<false, false><<<gg, 32, 0, stream>>>(buf0, Wqf, bqf, nullptr, buf3);
    final_attn<<<BATCH, 256, 0, stream>>>(buf3, K_att, mask, out);
}
